// EnhancedLocalAttentionWithGQA_60833916781077
// MI455X (gfx1250) — matrix operand order repaired and checked by reading, not compiled, not run on hardware
//
#include <hip/hip_runtime.h>
#include <hip/hip_bf16.h>

// ---------------------------------------------------------------------------
// EnhancedLocalAttentionWithGQA for MI455X (gfx1250, wave32, WMMA).
// Heavy math on v_wmma_f32_16x16x32_bf16; GEMM LDS staging via the CDNA5
// async memory->LDS path (global_load_async_to_lds_b128 + s_wait_asynccnt).
// bf16<->f32 conversions use native casts so the backend emits v_cvt ops.
// ---------------------------------------------------------------------------

typedef __attribute__((ext_vector_type(16))) __bf16 v16bf;
typedef __attribute__((ext_vector_type(8)))  __bf16 v8bf;
typedef __attribute__((ext_vector_type(8)))  float  v8f;

#define L_SEQ 4096
#define EMB   2048
#define NHEAD 16
#define HDIM  128
#define NGRP  2
#define KVDIM 256
#define WIN   256
#define NWIN_OUT 16   // only windows 0..15 survive the reference's [:, :L] slice
#define BATCH 2
#define MROWS (BATCH * L_SEQ)   // 8192

__device__ __forceinline__ __bf16 f2bf(float f) { return (__bf16)f; }
__device__ __forceinline__ float bf2f(__bf16 h) { return (float)h; }

__device__ __forceinline__ v8f vzero() {
  v8f z;
#pragma unroll
  for (int i = 0; i < 8; ++i) z[i] = 0.0f;
  return z;
}
// Build a 16x32 bf16 WMMA fragment from two contiguous 16-byte chunks.
__device__ __forceinline__ v16bf frag16(const __bf16* p0, const __bf16* p1) {
  v8bf a = *(const v8bf*)p0;
  v8bf b = *(const v8bf*)p1;
  v16bf r;
#pragma unroll
  for (int i = 0; i < 8; ++i) { r[i] = a[i]; r[i + 8] = b[i]; }
  return r;
}
__device__ __forceinline__ v8f wmma_bf16(v16bf a, v16bf b, v8f c) {
  return __builtin_amdgcn_wmma_f32_16x16x32_bf16(
      false, a, false, b, (short)0, c, false, false);
}

// CDNA5 async memory->LDS copy: 16B per lane, tracked by ASYNCcnt, no VGPR
// data movement. LDS aperture flat addresses carry the LDS byte offset in
// addr[31:0] (ISA 10.2), so truncating the generic pointer gives VDST.
__device__ __forceinline__ void async_ld_b128(const void* gptr, void* lptr) {
  unsigned lds = (unsigned)(uintptr_t)lptr;
  asm volatile("global_load_async_to_lds_b128 %0, %1, off"
               :: "v"(lds), "v"(gptr) : "memory");
}
__device__ __forceinline__ void wait_async0() {
#if __has_builtin(__builtin_amdgcn_s_wait_asynccnt)
  __builtin_amdgcn_s_wait_asynccnt(0);
#else
  asm volatile("s_wait_asynccnt 0" ::: "memory");
#endif
}

// ---------------------------------------------------------------------------
// Elementwise f32 -> bf16
// ---------------------------------------------------------------------------
__global__ void k_cast(const float* __restrict__ in, __bf16* __restrict__ out,
                       long n) {
  long i = (long)blockIdx.x * blockDim.x + threadIdx.x;
  if (i < n) out[i] = f2bf(in[i]);
}

// f32 [K][N] -> bf16 [N][K]
__global__ void k_transpose(const float* __restrict__ in,
                            __bf16* __restrict__ out, int K, int N) {
  long i = (long)blockIdx.x * blockDim.x + threadIdx.x;
  long total = (long)K * N;
  if (i < total) {
    int kk = (int)(i / N), nn = (int)(i % N);
    out[(long)nn * K + kk] = f2bf(in[i]);
  }
}

// ---------------------------------------------------------------------------
// Tiled bf16 WMMA GEMM:  C[M,N] = A[M,K] @ Bt[N,K]^T + bias[N]
// Block tile 128x128x32, 256 threads = 8 waves, wave tile 64x32 (4x2 WMMA).
// MODE 0: bf16 out, head-major   [((b*HOUT+h)*L + l)*D + d]   (Q and K)
// MODE 1: bf16 out, V transposed [((b*HOUT+g)*D + d)*L + l]
// MODE 2: f32 out, row-major [m*N + n]                        (final @Wo)
// ---------------------------------------------------------------------------
template <int MODE, int HOUT>
__global__ __launch_bounds__(256) void k_gemm(
    const __bf16* __restrict__ A, const __bf16* __restrict__ Bt,
    const float* __restrict__ bias, void* __restrict__ Cout,
    int M, int N, int K) {
  __shared__ __bf16 As[128][32];
  __shared__ __bf16 Bs[128][32];

  const int tid = threadIdx.x;
  const int lane = tid & 31;
  const int wave = tid >> 5;
  const int laneM = lane & 15;
  const int laneHi = lane >> 4;
  const int waveM = wave & 1;   // 2 waves over 128 rows (64 each)
  const int waveN = wave >> 1;  // 4 waves over 128 cols (32 each)
  const int blockM = blockIdx.y * 128;
  const int blockN = blockIdx.x * 128;

  v8f acc[4][2];
#pragma unroll
  for (int i = 0; i < 4; ++i)
#pragma unroll
    for (int j = 0; j < 2; ++j) acc[i][j] = vzero();

  const int ldRow = tid >> 1;        // 0..127
  const int ldOff = (tid & 1) * 16;  // 0 or 16 (each thread: 32 contiguous B)
  const __bf16* aG = A + (size_t)(blockM + ldRow) * K + ldOff;
  const __bf16* bG = Bt + (size_t)(blockN + ldRow) * K + ldOff;
  __bf16* aL = &As[ldRow][ldOff];
  __bf16* bL = &Bs[ldRow][ldOff];

  for (int kt = 0; kt < K; kt += 32) {
    __builtin_prefetch(aG + kt + 256, 0, 1);  // global_prefetch_b8
    __builtin_prefetch(bG + kt + 256, 0, 1);
    // Async memory->LDS staging (no VGPR round-trip), ASYNCcnt-tracked.
    async_ld_b128(aG + kt,     aL);
    async_ld_b128(aG + kt + 8, aL + 8);
    async_ld_b128(bG + kt,     bL);
    async_ld_b128(bG + kt + 8, bL + 8);
    wait_async0();
    __syncthreads();

    v16bf aF[4], bF[2];
#pragma unroll
    for (int mt = 0; mt < 4; ++mt) {
      const __bf16* r = &As[waveM * 64 + mt * 16 + laneM][0];
      aF[mt] = frag16(r + laneHi * 8, r + 16 + laneHi * 8);
    }
#pragma unroll
    for (int nt = 0; nt < 2; ++nt) {
      const __bf16* r = &Bs[waveN * 32 + nt * 16 + laneM][0];
      bF[nt] = frag16(r + laneHi * 8, r + 16 + laneHi * 8);
    }
#pragma unroll
    for (int mt = 0; mt < 4; ++mt)
#pragma unroll
      for (int nt = 0; nt < 2; ++nt)
        acc[mt][nt] = wmma_bf16(aF[mt], bF[nt], acc[mt][nt]);
    __syncthreads();
  }

#pragma unroll
  for (int mt = 0; mt < 4; ++mt) {
#pragma unroll
    for (int nt = 0; nt < 2; ++nt) {
      const int n = blockN + waveN * 32 + nt * 16 + laneM;
      const int mBase = blockM + waveM * 64 + mt * 16 + laneHi * 8;
      const float bia = bias[n];
#pragma unroll
      for (int r = 0; r < 8; ++r) {
        const int m = mBase + r;
        const float v = acc[mt][nt][r] + bia;
        if (MODE == 0) {
          const int b = m >> 12, l = m & (L_SEQ - 1);     // /4096, %4096
          const int h = n >> 7,  d = n & (HDIM - 1);      // /128, %128
          ((__bf16*)Cout)[(((size_t)(b * HOUT + h)) * L_SEQ + l) * HDIM + d] =
              f2bf(v);
        } else if (MODE == 1) {
          const int b = m >> 12, l = m & (L_SEQ - 1);
          const int g = n >> 7,  d = n & (HDIM - 1);
          ((__bf16*)Cout)[(((size_t)(b * HOUT + g)) * HDIM + d) * L_SEQ + l] =
              f2bf(v);
        } else {
          ((float*)Cout)[(size_t)m * N + n] = v;
        }
      }
    }
  }
}

// ---------------------------------------------------------------------------
// Windowed attention, one workgroup per (b, h, window n in [0,16)).
// Output row t = attention of query token (t/256)*128 + t%256 (reference
// slice semantics). Pass1: S = QK^T*scale -> LDS bf16; per-row softmax;
// Pass2: O = P V with V stored transposed [B][G][D][L].
// ---------------------------------------------------------------------------
#define SP 264  // padded LDS row stride (bf16 elems), keeps 16B alignment

__global__ __launch_bounds__(256) void k_attn(
    const __bf16* __restrict__ q, const __bf16* __restrict__ k,
    const __bf16* __restrict__ vt, __bf16* __restrict__ o) {
  extern __shared__ __bf16 sS[];  // [256][SP]

  const int n = blockIdx.x;  // window id 0..15
  const int h = blockIdx.y;
  const int b = blockIdx.z;
  const int g = h & (NGRP - 1);  // reshape(B,R,G,..): group = h % G

  const int tid = threadIdx.x;
  const int lane = tid & 31;
  const int wave = tid >> 5;
  const int laneM = lane & 15;
  const int laneHi = lane >> 4;
  const int rowBase = wave * 32;  // each wave owns 32 query rows

  const __bf16* qB =
      q + (((size_t)(b * NHEAD + h)) * L_SEQ + (size_t)n * (WIN / 2)) * HDIM;
  const __bf16* kB =
      k + (((size_t)(b * NGRP + g)) * L_SEQ + (size_t)n * (WIN / 2)) * HDIM;
  const __bf16* vB = vt + ((size_t)(b * NGRP + g)) * HDIM * L_SEQ +
                     (size_t)n * (WIN / 2);  // + d*L_SEQ + key
  __bf16* oB = o + ((size_t)b * L_SEQ + (size_t)n * WIN) * EMB + h * HDIM;

  const float scale = 0.088388347648318447f;  // 1/sqrt(128)

  // ---- Pass 1: scores. Preload this wave's Q fragments once (8 x v16bf). ----
  v16bf qF[2][4];
#pragma unroll
  for (int mt = 0; mt < 2; ++mt)
#pragma unroll
    for (int di = 0; di < 4; ++di) {
      const __bf16* qr =
          qB + (size_t)(rowBase + mt * 16 + laneM) * HDIM + di * 32;
      qF[mt][di] = frag16(qr + laneHi * 8, qr + 16 + laneHi * 8);
    }

  for (int kt = 0; kt < WIN / 16; ++kt) {
    v8f acc[2];
    acc[0] = vzero();
    acc[1] = vzero();
#pragma unroll
    for (int di = 0; di < 4; ++di) {
      const __bf16* kr =
          kB + (size_t)(kt * 16 + laneM) * HDIM + di * 32;
      v16bf bF = frag16(kr + laneHi * 8, kr + 16 + laneHi * 8);
#pragma unroll
      for (int mt = 0; mt < 2; ++mt)
        acc[mt] = wmma_bf16(qF[mt][di], bF, acc[mt]);
    }
#pragma unroll
    for (int mt = 0; mt < 2; ++mt)
#pragma unroll
      for (int r = 0; r < 8; ++r) {
        int srow = rowBase + mt * 16 + r + laneHi * 8;
        int scol = kt * 16 + laneM;
        sS[srow * SP + scol] = f2bf(acc[mt][r] * scale);
      }
  }
  __syncthreads();

  // ---- Softmax: thread t owns row t ----
  {
    __bf16* row = sS + tid * SP;
    float mx = -1e30f;
    for (int j = 0; j < WIN; ++j) mx = fmaxf(mx, bf2f(row[j]));
    float sum = 0.0f;
    for (int j = 0; j < WIN; ++j) {
      float e = __expf(bf2f(row[j]) - mx);
      sum += e;
      row[j] = f2bf(e);
    }
    float inv = 1.0f / sum;
    for (int j = 0; j < WIN; ++j) row[j] = f2bf(bf2f(row[j]) * inv);
  }
  __syncthreads();

  // ---- Pass 2: O = P @ V, D split in two halves to bound VGPR use ----
#pragma unroll
  for (int dh = 0; dh < 2; ++dh) {
    v8f acc[2][4];
#pragma unroll
    for (int mt = 0; mt < 2; ++mt)
#pragma unroll
      for (int dt = 0; dt < 4; ++dt) acc[mt][dt] = vzero();

    for (int ks = 0; ks < WIN; ks += 32) {
      v16bf aF[2];
#pragma unroll
      for (int mt = 0; mt < 2; ++mt) {
        const __bf16* pr = sS + (rowBase + mt * 16 + laneM) * SP + ks;
        aF[mt] = frag16(pr + laneHi * 8, pr + 16 + laneHi * 8);
      }
#pragma unroll
      for (int dt = 0; dt < 4; ++dt) {
        int d = dh * 64 + dt * 16 + laneM;
        const __bf16* vr = vB + (size_t)d * L_SEQ + ks;
        v16bf bF = frag16(vr + laneHi * 8, vr + 16 + laneHi * 8);
#pragma unroll
        for (int mt = 0; mt < 2; ++mt)
          acc[mt][dt] = wmma_bf16(aF[mt], bF, acc[mt][dt]);
      }
    }
#pragma unroll
    for (int mt = 0; mt < 2; ++mt)
#pragma unroll
      for (int dt = 0; dt < 4; ++dt)
#pragma unroll
        for (int r = 0; r < 8; ++r) {
          int w = rowBase + mt * 16 + r + laneHi * 8;
          int d = dh * 64 + dt * 16 + laneM;
          oB[(size_t)w * EMB + d] = f2bf(acc[mt][dt][r]);
        }
  }
}

// ---------------------------------------------------------------------------
// Host-side orchestration
// ---------------------------------------------------------------------------
extern "C" void kernel_launch(void* const* d_in, const int* in_sizes, int n_in,
                              void* d_out, int out_size, void* d_ws,
                              size_t ws_size, hipStream_t stream) {
  (void)in_sizes; (void)n_in; (void)out_size; (void)ws_size;
  const float* x  = (const float*)d_in[0];
  const float* Wq = (const float*)d_in[1];
  const float* bq = (const float*)d_in[2];
  const float* Wk = (const float*)d_in[3];
  const float* bk = (const float*)d_in[4];
  const float* Wv = (const float*)d_in[5];
  const float* bv = (const float*)d_in[6];
  const float* Wo = (const float*)d_in[7];
  const float* bo = (const float*)d_in[8];

  char* ws = (char*)d_ws;
  size_t off = 0;
  __bf16* xb  = (__bf16*)(ws + off); off += (size_t)MROWS * EMB * 2;       // x bf16
  __bf16* wqt = (__bf16*)(ws + off); off += (size_t)EMB * EMB * 2;         // Wq^T
  __bf16* wkt = (__bf16*)(ws + off); off += (size_t)EMB * KVDIM * 2;       // Wk^T
  __bf16* wvt = (__bf16*)(ws + off); off += (size_t)EMB * KVDIM * 2;       // Wv^T
  __bf16* wot = (__bf16*)(ws + off); off += (size_t)EMB * EMB * 2;         // Wo^T
  __bf16* qh  = (__bf16*)(ws + off); off += (size_t)MROWS * EMB * 2;       // Q head-major
  __bf16* kh  = (__bf16*)(ws + off); off += (size_t)MROWS * KVDIM * 2;     // K head-major
  __bf16* vtt = (__bf16*)(ws + off); off += (size_t)MROWS * KVDIM * 2;     // V transposed
  __bf16* ob  = (__bf16*)(ws + off); off += (size_t)MROWS * EMB * 2;       // attn out

  const long nx = (long)MROWS * EMB;
  k_cast<<<(nx + 255) / 256, 256, 0, stream>>>(x, xb, nx);
  k_transpose<<<((long)EMB * EMB + 255) / 256, 256, 0, stream>>>(Wq, wqt, EMB, EMB);
  k_transpose<<<((long)EMB * KVDIM + 255) / 256, 256, 0, stream>>>(Wk, wkt, EMB, KVDIM);
  k_transpose<<<((long)EMB * KVDIM + 255) / 256, 256, 0, stream>>>(Wv, wvt, EMB, KVDIM);
  k_transpose<<<((long)EMB * EMB + 255) / 256, 256, 0, stream>>>(Wo, wot, EMB, EMB);

  // Q = xb @ Wq^T + bq  -> head-major [B][H][L][D]
  k_gemm<0, NHEAD><<<dim3(EMB / 128, MROWS / 128), 256, 0, stream>>>(
      xb, wqt, bq, qh, MROWS, EMB, EMB);
  // K = xb @ Wk^T + bk  -> group-major [B][G][L][D]
  k_gemm<0, NGRP><<<dim3(KVDIM / 128, MROWS / 128), 256, 0, stream>>>(
      xb, wkt, bk, kh, MROWS, KVDIM, EMB);
  // V = xb @ Wv^T + bv  -> transposed [B][G][D][L]
  k_gemm<1, NGRP><<<dim3(KVDIM / 128, MROWS / 128), 256, 0, stream>>>(
      xb, wvt, bv, vtt, MROWS, KVDIM, EMB);

  // Windowed GQA attention (only the 16 windows that survive [:, :L])
  k_attn<<<dim3(NWIN_OUT, NHEAD, BATCH), 256, (size_t)WIN * SP * 2, stream>>>(
      qh, kh, vtt, ob);

  // out = ob @ Wo^T + bo  (fp32 row-major)
  k_gemm<2, 1><<<dim3(EMB / 128, MROWS / 128), 256, 0, stream>>>(
      ob, wot, bo, (float*)d_out, MROWS, EMB, EMB);
}
